// VisionHGAT_86423331930530
// MI455X (gfx1250) — matrix operand order repaired and checked by reading, not compiled, not run on hardware
//
#include <hip/hip_runtime.h>
#include <hip/hip_bf16.h>
#include <math.h>

// ---------------------------------------------------------------------------
// VisionHGAT for MI455X (gfx1250, wave32).
// GEMMs: f16 operands (pre-converted/padded/transposed), f32 accumulate via
// v_wmma_f32_16x16x32_f16; LDS-staged tiles, b128 loads, 2 WMMA per barrier.
// Edge softmax via u32-key atomicMax + f32 atomicAdd on L2-resident tables.
// Layer-2 bb/pb relations are dead code (output needs only xp) -> skipped.
// ---------------------------------------------------------------------------

typedef __attribute__((ext_vector_type(16))) _Float16 v16h;
typedef __attribute__((ext_vector_type(8)))  _Float16 v8h;
typedef __attribute__((ext_vector_type(8)))  float    v8f;

// -------------------------- helpers ----------------------------------------

__device__ __forceinline__ unsigned fkey(float f) {
  unsigned u = __float_as_uint(f);
  return (u & 0x80000000u) ? ~u : (u | 0x80000000u);  // monotone float->u32
}
__device__ __forceinline__ float fkey_inv(unsigned k) {
  unsigned u = (k & 0x80000000u) ? (k & 0x7FFFFFFFu) : ~k;
  return __uint_as_float(u);
}

__global__ void fill_u32(unsigned* __restrict__ p, unsigned v, size_t n) {
  size_t i = (size_t)blockIdx.x * blockDim.x + threadIdx.x;
  if (i < n) p[i] = v;
}

__global__ void init_bias2(float* __restrict__ out, const float* __restrict__ b1,
                           const float* __restrict__ b2, size_t rows, int dim) {
  size_t i = (size_t)blockIdx.x * blockDim.x + threadIdx.x;
  if (i < rows * (size_t)dim) {
    int c = (int)(i % dim);
    out[i] = b1[c] + b2[c];
  }
}

__global__ void build_self_edges(const int* __restrict__ s, const int* __restrict__ d,
                                 int E, int Nnode, int* __restrict__ os, int* __restrict__ od) {
  int i = blockIdx.x * blockDim.x + threadIdx.x;
  if (i >= E + Nnode) return;
  if (i < E) { os[i] = s[i]; od[i] = d[i]; }
  else       { os[i] = i - E; od[i] = i - E; }
}

// W[K,N] fp32 -> Wt[N,Kp] f16 (transpose + zero-pad K to Kp)
__global__ void prep_w(const float* __restrict__ W, _Float16* __restrict__ Wt,
                       int K, int N, int Kp) {
  int i = blockIdx.x * blockDim.x + threadIdx.x;
  if (i >= N * Kp) return;
  int k = i % Kp, n = i / Kp;
  Wt[i] = (k < K) ? (_Float16)W[(unsigned)k * N + n] : (_Float16)0.0f;
}

// A[M,K] fp32 -> Ah[M,Kp] f16 (zero-pad K to Kp)
__global__ void prep_a(const float* __restrict__ A, _Float16* __restrict__ Ah,
                       int M, int K, int Kp) {
  int i = blockIdx.x * blockDim.x + threadIdx.x;
  if (i >= M * Kp) return;
  int k = i % Kp, m = i / Kp;
  Ah[i] = (k < K) ? (_Float16)A[(unsigned)m * K + k] : (_Float16)0.0f;
}

__global__ void cvt_f32_f16(const float* __restrict__ s, _Float16* __restrict__ d, size_t n) {
  size_t i = (size_t)blockIdx.x * blockDim.x + threadIdx.x;
  if (i < n) d[i] = (_Float16)s[i];
}

// -------------------------- WMMA GEMM --------------------------------------
// C[m, coloff+n] = act(A[M,Kp] @ Wt[N,Kp]^T + bias[n]);  A,Wt,C f16, bias f32.
// Kp % 32 == 0. grid.x=M/16, grid.y=ceil(N/128), block=256 (8 waves, one
// 16x16 C tile each). KCHUNK=64 staged in LDS, 2 WMMAs per barrier round.
// Fragment layouts (ISA 7.12.2):
//   A lane m=L%16:  K(e) = (e<8?e:e+8) + 8*(L>=16)  -> two 16B LDS chunks
//   B lane n=L%16:  K(e) = (e<8?e:e+8) + 8*(L>=16)  -> two 16B LDS chunks
//   C f32 VGPR r -> row r + 8*(L>=16), col L%16
#define LDS_STRIDE 72
__global__ __launch_bounds__(256)
void gemm_h(const _Float16* __restrict__ A, const _Float16* __restrict__ Wt,
            const float* __restrict__ bias, _Float16* __restrict__ C,
            int M, int N, int Kp, int ldc, int coloff, int act) {
  __shared__ _Float16 As[16 * LDS_STRIDE];
  __shared__ _Float16 Ws[128 * LDS_STRIDE];
  const int tid = threadIdx.x, lane = tid & 31, wv = tid >> 5;
  const int hi = lane >> 4;
  const int m0 = blockIdx.x * 16;
  const int nc0 = blockIdx.y * 128;
  const int ncols = (N - nc0 < 128) ? (N - nc0) : 128;
  const bool active = ((blockIdx.y * 8 + wv) * 16) < N;
  const int nl = (wv << 4) + (lane & 15);  // local col 0..127
  const int n = nc0 + nl;

  v8f acc = {};
  for (int k0 = 0; k0 < Kp; k0 += 64) {
    const int kc = (Kp - k0 < 64) ? (Kp - k0) : 64;  // 64 or 32
    if (tid < 128) {  // A tile: 16 rows x 64 halves (zero tail)
      int m = tid >> 3, kq = (tid & 7) * 8;
      v8h v = {};
      if (kq < kc) v = *(const v8h*)(A + (unsigned)(m0 + m) * Kp + k0 + kq);
      *(v8h*)&As[m * LDS_STRIDE + kq] = v;
    }
    for (int idx = tid; idx < (ncols << 3); idx += 256) {  // W tile
      int nn = idx >> 3, kq = (idx & 7) * 8;
      v8h v = {};
      if (kq < kc) v = *(const v8h*)(Wt + (unsigned)(nc0 + nn) * Kp + k0 + kq);
      *(v8h*)&Ws[nn * LDS_STRIDE + kq] = v;
    }
    __syncthreads();

    const _Float16* ap = &As[(lane & 15) * LDS_STRIDE];
    const _Float16* bp = &Ws[(active ? nl : (lane & 15)) * LDS_STRIDE];
#pragma unroll
    for (int kk = 0; kk < 64; kk += 32) {
      v8h a0 = *(const v8h*)(ap + kk + hi * 8);
      v8h a1 = *(const v8h*)(ap + kk + 16 + hi * 8);
      v8h b0 = *(const v8h*)(bp + kk + hi * 8);
      v8h b1 = *(const v8h*)(bp + kk + 16 + hi * 8);
      v16h a, b;
#pragma unroll
      for (int e = 0; e < 8; ++e) {
        a[e] = a0[e]; a[8 + e] = a1[e];
        b[e] = b0[e]; b[8 + e] = b1[e];
      }
      acc = __builtin_amdgcn_wmma_f32_16x16x32_f16(false, a, false, b,
                                                   (short)0, acc, false, false);
    }
    __syncthreads();
  }
  if (active) {
    float bv = bias ? bias[n] : 0.0f;
#pragma unroll
    for (int r = 0; r < 8; ++r) {
      float v = acc[r] + bv;
      if (act == 1) v = fmaxf(v, 0.0f);
      C[(unsigned)(m0 + r + hi * 8) * ldc + coloff + n] = (_Float16)v;
    }
  }
}

// -------------------------- 3D CNN (fp32) ----------------------------------
__global__ void conv3d_relu_pool(const float* __restrict__ in, const float* __restrict__ W,
                                 const float* __restrict__ b, float* __restrict__ out,
                                 int N, int IC, int OC, int D) {
  const int P = D >> 1;
  size_t idx = (size_t)blockIdx.x * blockDim.x + threadIdx.x;
  size_t total = (size_t)N * OC * P * P * P;
  if (idx >= total) return;
  int ow = (int)(idx % P); size_t t = idx / P;
  int oh = (int)(t % P); t /= P;
  int od = (int)(t % P); t /= P;
  int oc = (int)(t % OC); int n = (int)(t / OC);

  float acc[8];
  float bb = b[oc];
#pragma unroll
  for (int p = 0; p < 8; ++p) acc[p] = bb;

  const float* inn = in + (size_t)n * IC * D * D * D;
  const float* wcc = W + (size_t)oc * IC * 27;
  for (int ic = 0; ic < IC; ++ic) {
    const float* ip = inn + (size_t)ic * D * D * D;
    const float* wp = wcc + ic * 27;
    for (int kz = 0; kz < 3; ++kz)
      for (int ky = 0; ky < 3; ++ky)
        for (int kx = 0; kx < 3; ++kx) {
          float wvv = wp[(kz * 3 + ky) * 3 + kx];
#pragma unroll
          for (int p = 0; p < 8; ++p) {
            int z = od * 2 + (p >> 2)       + kz - 1;
            int y = oh * 2 + ((p >> 1) & 1) + ky - 1;
            int x = ow * 2 + (p & 1)        + kx - 1;
            if ((unsigned)z < (unsigned)D && (unsigned)y < (unsigned)D &&
                (unsigned)x < (unsigned)D)
              acc[p] += wvv * ip[((size_t)z * D + y) * D + x];
          }
        }
  }
  float m = acc[0];
#pragma unroll
  for (int p = 1; p < 8; ++p) m = fmaxf(m, acc[p]);
  out[idx] = fmaxf(m, 0.0f);  // max(relu) == relu(max)
}

// SAME 3x3x3 conv on 1x1x1 volume == matmul with kernel center; +ReLU
__global__ void conv_center_relu(const float* __restrict__ in, const float* __restrict__ W,
                                 const float* __restrict__ b, float* __restrict__ out,
                                 int N, int C) {
  int i = blockIdx.x * blockDim.x + threadIdx.x;
  if (i >= N * C) return;
  int oc = i % C, n = i / C;
  float acc = b[oc];
  for (int ic = 0; ic < C; ++ic)
    acc += W[((unsigned)oc * C + ic) * 27 + 13] * in[(unsigned)n * C + ic];
  out[i] = fmaxf(acc, 0.0f);
}

// -------------------------- GATv2 edge stages ------------------------------
__global__ __launch_bounds__(256)
void edge_score(const int* __restrict__ src, const int* __restrict__ dst, int E,
                const _Float16* __restrict__ xl, const _Float16* __restrict__ xr,
                const float* __restrict__ att, float* __restrict__ s,
                unsigned* __restrict__ mkey) {
  int e = blockIdx.x * 8 + (threadIdx.x >> 5);
  int lane = threadIdx.x & 31;
  if (e >= E) return;
  int si = src[e], di = dst[e];
  const _Float16* xls = xl + (size_t)si * 256;
  const _Float16* xrd = xr + (size_t)di * 256;
#pragma unroll
  for (int h = 0; h < 2; ++h) {
    float a = 0.0f;
    for (int c = lane; c < 128; c += 32) {
      float v = (float)xls[h * 128 + c] + (float)xrd[h * 128 + c];
      v = (v > 0.0f) ? v : 0.2f * v;  // LeakyReLU(0.2)
      a += att[h * 128 + c] * v;
    }
#pragma unroll
    for (int off = 16; off; off >>= 1) a += __shfl_xor(a, off, 32);
    if (lane == 0) {
      s[(size_t)e * 2 + h] = a;
      atomicMax(&mkey[(size_t)di * 2 + h], fkey(a));
    }
  }
}

__global__ void edge_exp(const int* __restrict__ dst, int E, float* __restrict__ s,
                         const unsigned* __restrict__ mkey, float* __restrict__ z) {
  size_t idx = (size_t)blockIdx.x * blockDim.x + threadIdx.x;
  if (idx >= (size_t)E * 2) return;
  int e = (int)(idx >> 1), h = (int)(idx & 1);
  int di = dst[e];
  float w = __expf(s[idx] - fkey_inv(mkey[(size_t)di * 2 + h]));
  s[idx] = w;
  atomicAdd(&z[(size_t)di * 2 + h], w);
}

__global__ __launch_bounds__(256)
void edge_aggr(const int* __restrict__ src, const int* __restrict__ dst, int E,
               const _Float16* __restrict__ xl, const float* __restrict__ w,
               const float* __restrict__ z, float* __restrict__ out) {
  int e = blockIdx.x * 8 + (threadIdx.x >> 5);
  int lane = threadIdx.x & 31;
  if (e >= E) return;
  int si = src[e], di = dst[e];
  float a0 = w[(size_t)e * 2 + 0] / (z[(size_t)di * 2 + 0] + 1e-16f);
  float a1 = w[(size_t)e * 2 + 1] / (z[(size_t)di * 2 + 1] + 1e-16f);
  const _Float16* xls = xl + (size_t)si * 256;
  float* od = out + (size_t)di * 256;
  for (int c = lane; c < 256; c += 32)
    atomicAdd(&od[c], (float)xls[c] * ((c < 128) ? a0 : a1));
}

__global__ void out_proj(const float* __restrict__ x, const float* __restrict__ W,
                         const float* __restrict__ b, float* __restrict__ out, int N) {
  int i = blockIdx.x * blockDim.x + threadIdx.x;
  if (i >= N) return;
  float acc = b[0];
  for (int k = 0; k < 256; ++k) acc += x[(unsigned)i * 256 + k] * W[k];
  out[i] = acc;
}

// ===========================================================================

extern "C" void kernel_launch(void* const* d_in, const int* in_sizes, int n_in,
                              void* d_out, int out_size, void* d_ws, size_t ws_size,
                              hipStream_t stream) {
  if (n_in < 71) return;

  constexpr int NB = 30000, NP = 1024;
  constexpr int EBB = 480000, EPP = 8192, EBP = 32768, EPB = 32768;
  constexpr int EBBs = EBB + NB, EPPs = EPP + NP;

  enum {
    I_XB = 0, I_XP, I_IMG,
    I_EBB_S, I_EBB_D, I_EPP_S, I_EPP_D, I_EBP_S, I_EBP_D, I_EPB_S, I_EPB_D,
    FB_W0, FB_W1, FB_W2, FB_B0, FB_B1, FB_B2,
    FP_W0, FP_W1, FP_W2, FP_B0, FP_B1, FP_B2,
    CN_W0, CN_W1, CN_W2, CN_W3, CN_W4, CN_W5,
    CN_B0, CN_B1, CN_B2, CN_B3, CN_B4, CN_B5, CN_FCW, CN_FCB,
    G1_BB_WL, G1_BB_WR, G1_BB_AT, G1_BB_BI,
    G1_PP_WL, G1_PP_WR, G1_PP_AT, G1_PP_BI,
    G1_BP_WL, G1_BP_WR, G1_BP_AT, G1_BP_BI,
    G1_PB_WL, G1_PB_WR, G1_PB_AT, G1_PB_BI,
    G2_BB_WL, G2_BB_WR, G2_BB_AT, G2_BB_BI,
    G2_PP_WL, G2_PP_WR, G2_PP_AT, G2_PP_BI,
    G2_BP_WL, G2_BP_WR, G2_BP_AT, G2_BP_BI,
    G2_PB_WL, G2_PB_WR, G2_PB_AT, G2_PB_BI,
    O_W, O_B
  };
  auto f32 = [&](int i) { return (const float*)d_in[i]; };
  auto i32 = [&](int i) { return (const int*)d_in[i]; };

  // ---- workspace carve-out (byte cursor, 16B aligned) ----
  char* base = (char*)d_ws;
  size_t off = 0;
  auto alloc = [&](size_t bytes) -> void* {
    void* p = base + off;
    off = (off + bytes + 15) & ~(size_t)15;
    return p;
  };
  auto AF = [&](size_t n) { return (float*)alloc(n * 4); };
  auto AH = [&](size_t n) { return (_Float16*)alloc(n * 2); };
  auto AI = [&](size_t n) { return (int*)alloc(n * 4); };

  _Float16* xbr_h = AH((size_t)NB * 32);
  _Float16* xpp_h = AH((size_t)NP * 64);
  _Float16* tmpA_h = AH((size_t)NB * 128);
  _Float16* tmpB_h = AH((size_t)NB * 128);
  _Float16* xb_h = AH((size_t)NB * 128);
  _Float16* xp_h = AH((size_t)NP * 128);
  float* p2 = AF((size_t)NP * 24 * 64);
  float* p3 = AF((size_t)NP * 24 * 8);
  float* p4 = AF((size_t)NP * 24);
  float* p5 = AF((size_t)NP * 24);
  float* p6 = AF((size_t)NP * 24);
  _Float16* p6h = AH((size_t)NP * 32);
  // 4 contiguous big f16 projections; CNN p1 (NP*24*512 f32 = 50.3MB) aliases
  // this 61.4MB span (written after p1 is consumed).
  _Float16* xl_bb = AH((size_t)NB * 256);
  _Float16* xr_bb = AH((size_t)NB * 256);
  _Float16* xl_bp = AH((size_t)NB * 256);
  _Float16* xr_pb = AH((size_t)NB * 256);
  _Float16* xl_pp = AH((size_t)NP * 256);
  _Float16* xr_pp = AH((size_t)NP * 256);
  _Float16* xl_pb = AH((size_t)NP * 256);
  _Float16* xr_bp = AH((size_t)NP * 256);
  float* ob = AF((size_t)NB * 256);
  float* op = AF((size_t)NP * 256);
  float* op2 = AF((size_t)NP * 256);
  _Float16* ob_h = AH((size_t)NB * 256);
  _Float16* op_h = AH((size_t)NP * 256);
  unsigned* m_bb = (unsigned*)AF((size_t)NB * 2);
  float* z_bb = AF((size_t)NB * 2);
  unsigned* m_pb = (unsigned*)AF((size_t)NB * 2);
  float* z_pb = AF((size_t)NB * 2);
  unsigned* m_pp = (unsigned*)AF((size_t)NP * 2);
  float* z_pp = AF((size_t)NP * 2);
  unsigned* m_bp = (unsigned*)AF((size_t)NP * 2);
  float* z_bp = AF((size_t)NP * 2);
  float* s_bb = AF((size_t)EBBs * 2);
  float* s_pp = AF((size_t)EPPs * 2);
  float* s_bp = AF((size_t)EBP * 2);
  float* s_pb = AF((size_t)EPB * 2);
  int* ebbS = AI(EBBs);
  int* ebbD = AI(EBBs);
  int* eppS = AI(EPPs);
  int* eppD = AI(EPPs);
  // transposed f16 weights [N, Kp]
  _Float16* wt_fb0 = AH(128 * 32);
  _Float16* wt_fb1 = AH(128 * 128);
  _Float16* wt_fb2 = AH(128 * 128);
  _Float16* wt_fp0 = AH(64 * 64);
  _Float16* wt_fp1 = AH(64 * 64);
  _Float16* wt_fp2 = AH(64 * 64);
  _Float16* wt_fc = AH(64 * 32);
  _Float16* wt_g1[8];
  for (int i = 0; i < 8; ++i) wt_g1[i] = AH(256 * 128);
  _Float16* wt_g2[4];
  for (int i = 0; i < 4; ++i) wt_g2[i] = AH(256 * 256);
  if (off > ws_size) return;  // insufficient scratch: no-op

  float* p1 = (float*)xl_bb;  // CNN alias, consumed before xl_* written
  _Float16* xl2_pp = xl_pp; _Float16* xr2_pp = xr_pp; _Float16* xr2_bp = xr_bp;
  _Float16* xl2_bp = xl_bb;

  auto gemm = [&](const _Float16* Ain, const _Float16* Wtin, const float* bin,
                  _Float16* Cout, int M, int N, int Kp, int ldc, int coloff, int act) {
    dim3 g(M / 16, (N + 127) / 128);
    gemm_h<<<g, 256, 0, stream>>>(Ain, Wtin, bin, Cout, M, N, Kp, ldc, coloff, act);
  };
  auto prepw = [&](int wi, _Float16* dst, int K, int N, int Kp) {
    prep_w<<<(N * Kp + 255) / 256, 256, 0, stream>>>(f32(wi), dst, K, N, Kp);
  };
  auto fill0 = [&](void* p, size_t n_u32) {
    fill_u32<<<(unsigned)((n_u32 + 255) / 256), 256, 0, stream>>>((unsigned*)p, 0u, n_u32);
  };
  auto relation = [&](const int* es, const int* ed, int E,
                      const _Float16* xl, const _Float16* xr, const float* att,
                      unsigned* mk, float* z, float* s, float* outbuf) {
    unsigned gb = (unsigned)((E + 7) / 8);
    edge_score<<<gb, 256, 0, stream>>>(es, ed, E, xl, xr, att, s, mk);
    edge_exp<<<(unsigned)(((size_t)E * 2 + 255) / 256), 256, 0, stream>>>(ed, E, s, mk, z);
    edge_aggr<<<gb, 256, 0, stream>>>(es, ed, E, xl, s, z, outbuf);
  };

  // ---- phase 0: operand prep + self-loop edge lists ----
  build_self_edges<<<(EBBs + 255) / 256, 256, 0, stream>>>(i32(I_EBB_S), i32(I_EBB_D), EBB, NB, ebbS, ebbD);
  build_self_edges<<<(EPPs + 255) / 256, 256, 0, stream>>>(i32(I_EPP_S), i32(I_EPP_D), EPP, NP, eppS, eppD);
  prep_a<<<(NB * 32 + 255) / 256, 256, 0, stream>>>(f32(I_XB), xbr_h, NB, 32, 32);
  prep_a<<<(NP * 64 + 255) / 256, 256, 0, stream>>>(f32(I_XP), xpp_h, NP, 34, 64);
  prepw(FB_W0, wt_fb0, 32, 128, 32);
  prepw(FB_W1, wt_fb1, 128, 128, 128);
  prepw(FB_W2, wt_fb2, 128, 128, 128);
  prepw(FP_W0, wt_fp0, 34, 64, 64);
  prepw(FP_W1, wt_fp1, 64, 64, 64);
  prepw(FP_W2, wt_fp2, 64, 64, 64);
  prepw(CN_FCW, wt_fc, 24, 64, 32);
  { int wl[8] = {G1_BB_WL, G1_BB_WR, G1_BP_WL, G1_PB_WR, G1_PP_WL, G1_PP_WR, G1_PB_WL, G1_BP_WR};
    for (int i = 0; i < 8; ++i) prepw(wl[i], wt_g1[i], 128, 256, 128); }
  { int wl[4] = {G2_PP_WL, G2_PP_WR, G2_BP_WR, G2_BP_WL};
    for (int i = 0; i < 4; ++i) prepw(wl[i], wt_g2[i], 256, 256, 256); }

  // ---- phase 1: ffn_branch [NB,32]->128->128->128 ----
  gemm(xbr_h, wt_fb0, f32(FB_B0), tmpA_h, NB, 128, 32, 128, 0, 1);
  gemm(tmpA_h, wt_fb1, f32(FB_B1), tmpB_h, NB, 128, 128, 128, 0, 1);
  gemm(tmpB_h, wt_fb2, f32(FB_B2), xb_h, NB, 128, 128, 128, 0, 0);

  // ---- phase 2: ffn_proposal [NP,34]->64->64->64 -> xp[:,0:64] ----
  gemm(xpp_h, wt_fp0, f32(FP_B0), tmpA_h, NP, 64, 64, 64, 0, 1);
  gemm(tmpA_h, wt_fp1, f32(FP_B1), tmpB_h, NP, 64, 64, 64, 0, 1);
  gemm(tmpB_h, wt_fp2, f32(FP_B2), xp_h, NP, 64, 64, 128, 0, 0);

  // ---- phase 3: CNN 3D -> xp[:,64:128] ----
  {
    size_t t1 = (size_t)NP * 24 * 512, t2 = (size_t)NP * 24 * 64,
           t3 = (size_t)NP * 24 * 8, t4 = (size_t)NP * 24;
    conv3d_relu_pool<<<(unsigned)((t1 + 255) / 256), 256, 0, stream>>>(f32(I_IMG), f32(CN_W0), f32(CN_B0), p1, NP, 2, 24, 16);
    conv3d_relu_pool<<<(unsigned)((t2 + 255) / 256), 256, 0, stream>>>(p1, f32(CN_W1), f32(CN_B1), p2, NP, 24, 24, 8);
    conv3d_relu_pool<<<(unsigned)((t3 + 255) / 256), 256, 0, stream>>>(p2, f32(CN_W2), f32(CN_B2), p3, NP, 24, 24, 4);
    conv3d_relu_pool<<<(unsigned)((t4 + 255) / 256), 256, 0, stream>>>(p3, f32(CN_W3), f32(CN_B3), p4, NP, 24, 24, 2);
    conv_center_relu<<<(unsigned)((t4 + 255) / 256), 256, 0, stream>>>(p4, f32(CN_W4), f32(CN_B4), p5, NP, 24);
    conv_center_relu<<<(unsigned)((t4 + 255) / 256), 256, 0, stream>>>(p5, f32(CN_W5), f32(CN_B5), p6, NP, 24);
    prep_a<<<(NP * 32 + 255) / 256, 256, 0, stream>>>(p6, p6h, NP, 24, 32);
    gemm(p6h, wt_fc, f32(CN_FCB), xp_h, NP, 64, 32, 128, 64, 0);
  }

  // ---- phase 4: GAT layer 1 (all 4 relations) ----
  gemm(xb_h, wt_g1[0], nullptr, xl_bb, NB, 256, 128, 256, 0, 0);
  gemm(xb_h, wt_g1[1], nullptr, xr_bb, NB, 256, 128, 256, 0, 0);
  gemm(xb_h, wt_g1[2], nullptr, xl_bp, NB, 256, 128, 256, 0, 0);
  gemm(xb_h, wt_g1[3], nullptr, xr_pb, NB, 256, 128, 256, 0, 0);
  gemm(xp_h, wt_g1[4], nullptr, xl_pp, NP, 256, 128, 256, 0, 0);
  gemm(xp_h, wt_g1[5], nullptr, xr_pp, NP, 256, 128, 256, 0, 0);
  gemm(xp_h, wt_g1[6], nullptr, xl_pb, NP, 256, 128, 256, 0, 0);
  gemm(xp_h, wt_g1[7], nullptr, xr_bp, NP, 256, 128, 256, 0, 0);

  fill0(m_bb, (size_t)NB * 2); fill0(z_bb, (size_t)NB * 2);
  fill0(m_pb, (size_t)NB * 2); fill0(z_pb, (size_t)NB * 2);
  fill0(m_pp, (size_t)NP * 2); fill0(z_pp, (size_t)NP * 2);
  fill0(m_bp, (size_t)NP * 2); fill0(z_bp, (size_t)NP * 2);
  init_bias2<<<(unsigned)(((size_t)NB * 256 + 255) / 256), 256, 0, stream>>>(ob, f32(G1_BB_BI), f32(G1_PB_BI), NB, 256);
  init_bias2<<<(unsigned)(((size_t)NP * 256 + 255) / 256), 256, 0, stream>>>(op, f32(G1_PP_BI), f32(G1_BP_BI), NP, 256);

  relation(ebbS, ebbD, EBBs, xl_bb, xr_bb, f32(G1_BB_AT), m_bb, z_bb, s_bb, ob);
  relation(i32(I_EPB_S), i32(I_EPB_D), EPB, xl_pb, xr_pb, f32(G1_PB_AT), m_pb, z_pb, s_pb, ob);
  relation(eppS, eppD, EPPs, xl_pp, xr_pp, f32(G1_PP_AT), m_pp, z_pp, s_pp, op);
  relation(i32(I_EBP_S), i32(I_EBP_D), EBP, xl_bp, xr_bp, f32(G1_BP_AT), m_bp, z_bp, s_bp, op);

  // ---- phase 5: GAT layer 2 (only pp and bp feed the output) ----
  cvt_f32_f16<<<(unsigned)(((size_t)NB * 256 + 255) / 256), 256, 0, stream>>>(ob, ob_h, (size_t)NB * 256);
  cvt_f32_f16<<<(unsigned)(((size_t)NP * 256 + 255) / 256), 256, 0, stream>>>(op, op_h, (size_t)NP * 256);
  gemm(op_h, wt_g2[0], nullptr, xl2_pp, NP, 256, 256, 256, 0, 0);
  gemm(op_h, wt_g2[1], nullptr, xr2_pp, NP, 256, 256, 256, 0, 0);
  gemm(op_h, wt_g2[2], nullptr, xr2_bp, NP, 256, 256, 256, 0, 0);
  gemm(ob_h, wt_g2[3], nullptr, xl2_bp, NB, 256, 256, 256, 0, 0);

  fill0(m_pp, (size_t)NP * 2); fill0(z_pp, (size_t)NP * 2);
  fill0(m_bp, (size_t)NP * 2); fill0(z_bp, (size_t)NP * 2);
  init_bias2<<<(unsigned)(((size_t)NP * 256 + 255) / 256), 256, 0, stream>>>(op2, f32(G2_PP_BI), f32(G2_BP_BI), NP, 256);

  relation(eppS, eppD, EPPs, xl2_pp, xr2_pp, f32(G2_PP_AT), m_pp, z_pp, s_pp, op2);
  relation(i32(I_EBP_S), i32(I_EBP_D), EBP, xl2_bp, xr2_bp, f32(G2_BP_AT), m_bp, z_bp, s_bp, op2);

  // ---- phase 6: final linear [NP,256] @ [256,1] + b ----
  out_proj<<<(NP + 255) / 256, 256, 0, stream>>>(op2, f32(O_W), f32(O_B), (float*)d_out, NP);
}
